// ByteLLM_GrugV2_23433341567853
// MI455X (gfx1250) — compile-verified, matrix operand order checked by reading, not run on hardware
//
#include <hip/hip_runtime.h>
#include <hip/hip_bf16.h>
#include <math.h>

// ---- model constants (fixed by reference) ----
#define SEQ_L   2048
#define NBATCH  2
#define NM      (NBATCH * SEQ_L)   // 4096 rows

typedef __attribute__((ext_vector_type(16))) _Float16 v16h;
typedef __attribute__((ext_vector_type(8)))  float    v8f;
typedef __attribute__((ext_vector_type(4)))  unsigned int v4u;

union HFrag { v4u u[2]; v16h h; };

__device__ __forceinline__ v16h frag_load(const _Float16* base, int ld, int row, int k0, int hi) {
  const _Float16* p = base + (long)row * ld + k0 + (hi << 3);
  HFrag f;
  f.u[0] = *(const v4u*)(const void*)(p);
  f.u[1] = *(const v4u*)(const void*)(p + 16);
  return f.h;
}

// async copy 16 bytes global -> LDS (CDNA5; tracked by ASYNCcnt, in-order completion)
__device__ __forceinline__ void async_g2l_b128(unsigned lds_byte_off, unsigned long long gaddr) {
  asm volatile("global_load_async_to_lds_b128 %0, %1, off"
               :: "v"(lds_byte_off), "v"(gaddr) : "memory");
}
__device__ __forceinline__ void wait_async_le0() {
  asm volatile("s_wait_asynccnt 0x0" ::: "memory");
}
__device__ __forceinline__ void wait_async_le2() {
  asm volatile("s_wait_asynccnt 0x2" ::: "memory");
}

// C(M,N) f32 = A(M,K)f16 @ W(N,K)f16^T   (pure GEMM; bias/act applied separately)
// Block tile 128x128 (8 waves: 4 in M x 2 in N), wave tile 32x64.
// W tile (128 x 32 f16 = 8 KB) double-buffered in LDS via async global->LDS.
// Requires: M % 128 == 0, K % 32 == 0. N guarded only when GUARD_N.
template <bool GUARD_N>
__global__ __launch_bounds__(256) void k_gemm_f16(
    const _Float16* __restrict__ A, const _Float16* __restrict__ W,
    float* __restrict__ C, int N, int K, int lda, int ldw, int ldc) {
  __shared__ __align__(16) _Float16 bt[2 * 128 * 32];  // 2 x 8 KB, row-major ld = 32

  const int tid  = threadIdx.x;
  const int lane = tid & 31;
  const int wave = tid >> 5;
  const int lo = lane & 15, hi = lane >> 4;
  const int row0 = blockIdx.y * 128 + (wave >> 1) * 32;  // 4 waves over M
  const int col0 = blockIdx.x * 128 + (wave & 1) * 64;   // 2 waves over N

  // ---- staging: 512 16B-chunks (128 rows x 4) per tile, 2 per thread ----
  const int e0 = tid, e1 = tid + 256;
  int gn0 = blockIdx.x * 128 + (e0 >> 2);
  int gn1 = blockIdx.x * 128 + (e1 >> 2);
  if (GUARD_N) {  // clamp: garbage lands only in columns whose stores are skipped
    if (gn0 >= N) gn0 = N - 1;
    if (gn1 >= N) gn1 = N - 1;
  }
  unsigned long long ga0 = (unsigned long long)(W + (long)gn0 * ldw + (e0 & 3) * 8);
  unsigned long long ga1 = (unsigned long long)(W + (long)gn1 * ldw + (e1 & 3) * 8);
  const unsigned l0 = (unsigned)(unsigned long long)&bt[e0 * 8];
  const unsigned l1 = (unsigned)(unsigned long long)&bt[e1 * 8];

  const int ar0 = row0 + lo, ar1 = row0 + 16 + lo;
  const int bbase = (wave & 1) * 64 + lo;

  v8f acc[2][4];
#pragma unroll
  for (int i = 0; i < 2; ++i)
#pragma unroll
    for (int j = 0; j < 4; ++j)
#pragma unroll
      for (int r = 0; r < 8; ++r) acc[i][j][r] = 0.0f;

  // prologue: tile 0 -> buffer 0
  async_g2l_b128(l0, ga0);
  async_g2l_b128(l1, ga1);
  ga0 += 64; ga1 += 64;

  const int steps = K >> 5;
  for (int q = 0; q < steps; ++q) {
    const int cur = q & 1;

    v16h a0 = frag_load(A, lda, ar0, q * 32, hi);
    v16h a1 = frag_load(A, lda, ar1, q * 32, hi);

    if (q + 1 < steps) {
      const unsigned nbo = (cur ^ 1) ? 8192u : 0u;
      async_g2l_b128(l0 + nbo, ga0);
      async_g2l_b128(l1 + nbo, ga1);
      ga0 += 64; ga1 += 64;
      wait_async_le2();   // oldest tile (cur) landed; next tile may be in flight
    } else {
      wait_async_le0();
    }
    __syncthreads();      // all waves' current tile visible

    const _Float16* btc = bt + (cur ? 4096 : 0);
    v16h b0 = frag_load(btc, 32, bbase + 0,  0, hi);
    v16h b1 = frag_load(btc, 32, bbase + 16, 0, hi);
    v16h b2 = frag_load(btc, 32, bbase + 32, 0, hi);
    v16h b3 = frag_load(btc, 32, bbase + 48, 0, hi);

    acc[0][0] = __builtin_amdgcn_wmma_f32_16x16x32_f16(false, a0, false, b0, (short)0, acc[0][0], false, false);
    acc[0][1] = __builtin_amdgcn_wmma_f32_16x16x32_f16(false, a0, false, b1, (short)0, acc[0][1], false, false);
    acc[0][2] = __builtin_amdgcn_wmma_f32_16x16x32_f16(false, a0, false, b2, (short)0, acc[0][2], false, false);
    acc[0][3] = __builtin_amdgcn_wmma_f32_16x16x32_f16(false, a0, false, b3, (short)0, acc[0][3], false, false);
    acc[1][0] = __builtin_amdgcn_wmma_f32_16x16x32_f16(false, a1, false, b0, (short)0, acc[1][0], false, false);
    acc[1][1] = __builtin_amdgcn_wmma_f32_16x16x32_f16(false, a1, false, b1, (short)0, acc[1][1], false, false);
    acc[1][2] = __builtin_amdgcn_wmma_f32_16x16x32_f16(false, a1, false, b2, (short)0, acc[1][2], false, false);
    acc[1][3] = __builtin_amdgcn_wmma_f32_16x16x32_f16(false, a1, false, b3, (short)0, acc[1][3], false, false);

    __syncthreads();      // protect buffer re-issued next iteration
  }

#pragma unroll
  for (int j = 0; j < 4; ++j) {
    const int n = col0 + j * 16 + lo;
    if (GUARD_N && n >= N) continue;
#pragma unroll
    for (int i = 0; i < 2; ++i) {
      float* cp = C + (long)(row0 + i * 16 + hi * 8) * ldc + n;
#pragma unroll
      for (int r = 0; r < 8; ++r) cp[(long)r * ldc] = acc[i][j][r];
    }
  }
}

// x[i] = act(x[i] + bias[i & nmask]); act: 1=gelu(exact) 3=softplus
__global__ void k_bias_act(float* __restrict__ x, const float* __restrict__ bias,
                           int nmask, int total, int act) {
  int i = blockIdx.x * 256 + threadIdx.x;
  if (i >= total) return;
  float v = x[i] + bias[i & nmask];
  if (act == 1)      v = 0.5f * v * (1.0f + erff(v * 0.7071067811865476f));
  else if (act == 3) v = (v > 20.0f) ? v : log1pf(expf(v));
  x[i] = v;
}

__global__ void k_embed(const int* __restrict__ tok, const float* __restrict__ emb,
                        float* __restrict__ out, int total) {
  int i = blockIdx.x * 256 + threadIdx.x;
  if (i >= total) return;
  int m = i / 384, c = i % 384;
  out[i] = emb[(long)tok[m] * 384 + c];
}

__global__ void k_cvt(const float* __restrict__ src, _Float16* __restrict__ dst, int n) {
  int i = blockIdx.x * 256 + threadIdx.x;
  if (i < n) dst[i] = (_Float16)src[i];
}

__global__ void k_cvt_rows(const float* __restrict__ src, int sld, int cols,
                           _Float16* __restrict__ dst, int dld, int total) {
  int i = blockIdx.x * 256 + threadIdx.x;
  if (i >= total) return;
  int r = i / cols, c = i % cols;
  dst[(long)r * dld + c] = (_Float16)src[(long)r * sld + c];
}

// im2col: dst[m, c*KW+k] = src[b, t+k-pad, c] (dld = C*KW)
__global__ void k_im2col(const float* __restrict__ src, int sld, int KW, int pad,
                         _Float16* __restrict__ dst, int dld, int total) {
  int i = blockIdx.x * 256 + threadIdx.x;
  if (i >= total) return;
  int m = i / dld, j = i % dld;
  int c = j / KW, k = j % KW;
  int t = m & (SEQ_L - 1);
  int tt = t + k - pad;
  float v = 0.0f;
  if (tt >= 0 && tt < SEQ_L) v = src[(long)(m + tt - t) * sld + c];
  dst[i] = (_Float16)v;
}

__global__ __launch_bounds__(256) void k_ln512(const float* __restrict__ in,
    const float* __restrict__ g, const float* __restrict__ b,
    const float* __restrict__ pos, float* __restrict__ out) {
  __shared__ float ssum[256], ssq[256];
  const int row = blockIdx.x;
  const int tid = threadIdx.x;
  float v0 = in[(long)row * 512 + tid];
  float v1 = in[(long)row * 512 + 256 + tid];
  ssum[tid] = v0 + v1;
  ssq[tid]  = v0 * v0 + v1 * v1;
  __syncthreads();
  for (int s = 128; s > 0; s >>= 1) {
    if (tid < s) { ssum[tid] += ssum[tid + s]; ssq[tid] += ssq[tid + s]; }
    __syncthreads();
  }
  float mu  = ssum[0] * (1.0f / 512.0f);
  float var = ssq[0]  * (1.0f / 512.0f) - mu * mu;
  float rs  = rsqrtf(var + 1e-5f);
  int t = row & (SEQ_L - 1);
  float p0 = pos ? pos[(long)t * 512 + tid]       : 0.0f;
  float p1 = pos ? pos[(long)t * 512 + 256 + tid] : 0.0f;
  out[(long)row * 512 + tid]       = (v0 - mu) * rs * g[tid]       + b[tid]       + p0;
  out[(long)row * 512 + 256 + tid] = (v1 - mu) * rs * g[tid + 256] + b[tid + 256] + p1;
}

// depthwise causal conv (K=4) + SiLU over xi = xz[:, :1024]; writes f32 + f16
__global__ void k_dwconv_silu(const float* __restrict__ xz, const float* __restrict__ cw,
    const float* __restrict__ cb, float* __restrict__ xc, _Float16* __restrict__ xch,
    int total) {
  int i = blockIdx.x * 256 + threadIdx.x;
  if (i >= total) return;
  int d = i & 1023, m = i >> 10;
  int t = m & (SEQ_L - 1);
  float s = cb[d];
#pragma unroll
  for (int k = 0; k < 4; ++k) {
    int tt = t - 3 + k;
    if (tt >= 0) s += xz[(long)(m - 3 + k) * 2048 + d] * cw[d * 4 + k];
  }
  s = s / (1.0f + expf(-s));
  xc[i]  = s;
  xch[i] = (_Float16)s;
}

// selective scan: one thread per (b, d); h[16] in registers; fused *silu(z)
__global__ __launch_bounds__(256) void k_scan(
    const float* __restrict__ dt, const float* __restrict__ xc,
    const float* __restrict__ dbc, const float* __restrict__ xz,
    const float* __restrict__ A_log, const float* __restrict__ Dp,
    _Float16* __restrict__ yh) {
  int idx = blockIdx.x * 256 + threadIdx.x;  // 0..2047
  int d = idx & 1023, b = idx >> 10;
  float a[16], h[16];
#pragma unroll
  for (int s = 0; s < 16; ++s) { a[s] = -expf(A_log[d * 16 + s]); h[s] = 0.0f; }
  float Dd = Dp[d];
  long base = (long)b * SEQ_L;
  for (int t = 0; t < SEQ_L; ++t) {
    long m = base + t;
    float dtv = dt[m * 1024 + d];
    float xcv = xc[m * 1024 + d];
    float bx = dtv * xcv;
    const float* row = dbc + m * 64;
    float y = 0.0f;
#pragma unroll
    for (int s = 0; s < 16; ++s) {
      h[s] = expf(dtv * a[s]) * h[s] + bx * row[32 + s];
      y += h[s] * row[48 + s];
    }
    y += Dd * xcv;
    float z = xz[m * 2048 + 1024 + d];
    y *= z / (1.0f + expf(-z));
    yh[m * 1024 + d] = (_Float16)y;
  }
}

__global__ void k_head(const float* __restrict__ x, const float* __restrict__ fw,
                       const float* __restrict__ fb, float* __restrict__ out) {
  int i = blockIdx.x * 256 + threadIdx.x;  // 0..511
  if (i >= 512) return;
  int b = i >> 8, v = i & 255;
  const float* rep = x + ((long)b * SEQ_L + SEQ_L - 1) * 512;
  float s = fb[v];
  for (int k = 0; k < 512; ++k) s += rep[k] * fw[v * 512 + k];
  out[i] = s;
}

extern "C" void kernel_launch(void* const* d_in, const int* in_sizes, int n_in,
                              void* d_out, int out_size, void* d_ws, size_t ws_size,
                              hipStream_t stream) {
  (void)in_sizes; (void)n_in; (void)out_size; (void)ws_size;
  const int*   tokens   = (const int*)  d_in[0];
  const float* emb      = (const float*)d_in[1];
  const float* conv1_w  = (const float*)d_in[2];
  const float* conv1_b  = (const float*)d_in[3];
  const float* ln1_g    = (const float*)d_in[4];
  const float* ln1_b    = (const float*)d_in[5];
  const float* conv2_w  = (const float*)d_in[6];
  const float* conv2_b  = (const float*)d_in[7];
  const float* ln2_g    = (const float*)d_in[8];
  const float* ln2_b    = (const float*)d_in[9];
  const float* pos_emb  = (const float*)d_in[10];
  const float* in_proj  = (const float*)d_in[11];
  const float* conv_w   = (const float*)d_in[12];
  const float* conv_b   = (const float*)d_in[13];
  const float* x_proj   = (const float*)d_in[14];
  const float* dt_proj  = (const float*)d_in[15];
  const float* dt_bias  = (const float*)d_in[16];
  const float* A_log    = (const float*)d_in[17];
  const float* Dp       = (const float*)d_in[18];
  const float* out_proj = (const float*)d_in[19];
  const float* fc_w     = (const float*)d_in[20];
  const float* fc_b     = (const float*)d_in[21];

  const int M = NM;
  char* w = (char*)d_ws;
  float*    x_f32   = (float*)w;      w += (size_t)M * 512 * 4;    // layer activations
  float*    big_f32 = (float*)w;      w += (size_t)M * 2048 * 4;   // xz
  float*    xc_f32  = (float*)w;      w += (size_t)M * 1024 * 4;   // conv+silu out
  float*    dt_f32  = (float*)w;      w += (size_t)M * 1024 * 4;   // dt (also conv tmp)
  float*    dbc_f32 = (float*)w;      w += (size_t)M * 64 * 4;     // [dt_r | B | C]
  _Float16* a16     = (_Float16*)w;   w += (size_t)M * 2048 * 2;   // GEMM A staging
  _Float16* w16     = (_Float16*)w;   w += (size_t)2048 * 512 * 2; // GEMM W staging

  auto gemm = [&](const _Float16* A, const _Float16* Wm, float* C, int N, int K) {
    dim3 grid((N + 127) / 128, M / 128);
    if ((N & 127) == 0)
      k_gemm_f16<false><<<grid, 256, 0, stream>>>(A, Wm, C, N, K, K, K, N);
    else
      k_gemm_f16<true ><<<grid, 256, 0, stream>>>(A, Wm, C, N, K, K, K, N);
  };
  auto blk = [](int n) { return (n + 255) / 256; };

  // ---------------- front end ----------------
  k_embed<<<blk(M * 384), 256, 0, stream>>>(tokens, emb, x_f32, M * 384);
  k_im2col<<<blk(M * 1920), 256, 0, stream>>>(x_f32, 384, 5, 2, a16, 1920, M * 1920);
  k_cvt<<<blk(512 * 1920), 256, 0, stream>>>(conv1_w, w16, 512 * 1920);
  gemm(a16, w16, dt_f32, 512, 1920);
  k_bias_act<<<blk(M * 512), 256, 0, stream>>>(dt_f32, conv1_b, 511, M * 512, 1);
  k_ln512<<<M, 256, 0, stream>>>(dt_f32, ln1_g, ln1_b, nullptr, x_f32);
  k_im2col<<<blk(M * 1536), 256, 0, stream>>>(x_f32, 512, 3, 1, a16, 1536, M * 1536);
  k_cvt<<<blk(512 * 1536), 256, 0, stream>>>(conv2_w, w16, 512 * 1536);
  gemm(a16, w16, dt_f32, 512, 1536);
  k_bias_act<<<blk(M * 512), 256, 0, stream>>>(dt_f32, conv2_b, 511, M * 512, 1);
  k_ln512<<<M, 256, 0, stream>>>(dt_f32, ln2_g, ln2_b, pos_emb, x_f32);

  // ---------------- mamba layers ----------------
  for (int i = 0; i < 8; ++i) {
    k_cvt<<<blk(M * 512), 256, 0, stream>>>(x_f32, a16, M * 512);
    k_cvt<<<blk(2048 * 512), 256, 0, stream>>>(in_proj + (size_t)i * 2048 * 512, w16, 2048 * 512);
    gemm(a16, w16, big_f32, 2048, 512);                                   // xz

    k_dwconv_silu<<<blk(M * 1024), 256, 0, stream>>>(big_f32,
        conv_w + (size_t)i * 4096, conv_b + (size_t)i * 1024, xc_f32, a16, M * 1024);

    k_cvt<<<blk(64 * 1024), 256, 0, stream>>>(x_proj + (size_t)i * 64 * 1024, w16, 64 * 1024);
    gemm(a16, w16, dbc_f32, 64, 1024);                                    // dbc

    k_cvt_rows<<<blk(M * 32), 256, 0, stream>>>(dbc_f32, 64, 32, a16, 32, M * 32);
    k_cvt<<<blk(1024 * 32), 256, 0, stream>>>(dt_proj + (size_t)i * 1024 * 32, w16, 1024 * 32);
    gemm(a16, w16, dt_f32, 1024, 32);
    k_bias_act<<<blk(M * 1024), 256, 0, stream>>>(dt_f32, dt_bias + (size_t)i * 1024, 1023, M * 1024, 3);

    k_scan<<<2048 / 256, 256, 0, stream>>>(dt_f32, xc_f32, dbc_f32, big_f32,
        A_log + (size_t)i * 16384, Dp + (size_t)i * 1024, a16);           // y (f16)

    k_cvt<<<blk(512 * 1024), 256, 0, stream>>>(out_proj + (size_t)i * 512 * 1024, w16, 512 * 1024);
    gemm(a16, w16, x_f32, 512, 1024);                                     // next x
  }

  k_head<<<2, 256, 0, stream>>>(x_f32, fc_w, fc_b, (float*)d_out);
}
